// YOLOFv2_14723147891256
// MI455X (gfx1250) — compile-verified, matrix operand order checked by reading, not run on hardware
//
#include <hip/hip_runtime.h>
#include <hip/hip_bf16.h>
#include <math.h>

typedef __attribute__((ext_vector_type(16))) _Float16 v16h;
typedef __attribute__((ext_vector_type(8)))  float    v8f;
typedef __attribute__((ext_vector_type(4)))  int      v4i;

#define NUM_CLASSES 80
#define TOPK        1000
#define NPAD        1024
#define N_ANCHORS   262144
#define N_TOTAL     (N_ANCHORS * NUM_CLASSES)   /* 20,971,520 */
#define N4_TOTAL    (N_TOTAL / 4)
#define NBINS       8192
#define CAP         4096
#define NMS_THRESH  0.6f
#define CONF_THRESH 0.05f
#define MAX_COORD   4096.0f
#define AREA_SCALE  (1.0f / 1024.0f)
#define AREA_UNSCALE 1024.0f

/* ---- workspace layout (32-bit word offsets) ---- */
#define OFF_HIST 0
#define OFF_CNT  (NBINS)
#define OFF_TBIN (NBINS + 1)
#define OFF_CAND 8200                    /* byte offset 32800, 8B aligned */
#define OFF_TS   (OFF_CAND + 2 * CAP)    /* 16392: logits, later sigmoid  */
#define OFF_TI   (OFF_TS + NPAD)
#define OFF_BX1  (OFF_TI + NPAD)         /* 18440: 16B aligned; bx1..bar contiguous */
#define OFF_BY1  (OFF_BX1 + NPAD)
#define OFF_BX2  (OFF_BY1 + NPAD)
#define OFF_BY2  (OFF_BX2 + NPAD)
#define OFF_BAR  (OFF_BY2 + NPAD)       /* scaled areas */
#define OFF_VAL  (OFF_BAR + NPAD)
#define OFF_MASK (OFF_VAL + NPAD)       /* 1024 rows x 32 words */
#define WS_WORDS (OFF_MASK + NPAD * 32)

#if defined(__AMDGCN__) && __has_builtin(__builtin_amdgcn_global_load_async_to_lds_b128) && __has_builtin(__builtin_amdgcn_s_wait_asynccnt)
#define HAVE_ASYNC_LDS 1
typedef __attribute__((address_space(1))) v4i* as1_v4i_p;  /* global int4*  */
typedef __attribute__((address_space(3))) v4i* as3_v4i_p;  /* LDS int4*     */
#endif

__device__ __forceinline__ unsigned mono_key(unsigned u) {
  /* order-preserving float->uint map */
  return (u & 0x80000000u) ? ~u : (u | 0x80000000u);
}

/* ---------------- zero workspace ---------------- */
__global__ void k_zero(unsigned* __restrict__ ws, int n) {
  int stride = gridDim.x * blockDim.x;
  for (int i = blockIdx.x * blockDim.x + threadIdx.x; i < n; i += stride)
    ws[i] = 0u;
}

/* ---------------- pass A: histogram of monotonic keys ---------------- */
__global__ void k_hist(const uint4* __restrict__ p, unsigned* __restrict__ ghist, int n4) {
  __shared__ unsigned lh[NBINS];
  for (int b = threadIdx.x; b < NBINS; b += blockDim.x) lh[b] = 0u;
  __syncthreads();
  int stride = gridDim.x * blockDim.x;
  for (int i = blockIdx.x * blockDim.x + threadIdx.x; i < n4; i += stride) {
    if (i + stride < n4) __builtin_prefetch(p + i + stride, 0, 0);  /* global_prefetch_b8 */
    uint4 v = p[i];
    atomicAdd(&lh[mono_key(v.x) >> 19], 1u);
    atomicAdd(&lh[mono_key(v.y) >> 19], 1u);
    atomicAdd(&lh[mono_key(v.z) >> 19], 1u);
    atomicAdd(&lh[mono_key(v.w) >> 19], 1u);
  }
  __syncthreads();
  for (int b = threadIdx.x; b < NBINS; b += blockDim.x) {
    unsigned c = lh[b];
    if (c) atomicAdd(&ghist[b], c);
  }
}

/* ---------------- threshold bin (cumulative from top) ---------------- */
__global__ void k_thresh(const unsigned* __restrict__ hist, unsigned* __restrict__ tbin) {
  unsigned cum = 0;
  for (int b = NBINS - 1; b >= 0; --b) {
    cum += hist[b];
    if (cum >= (unsigned)TOPK) { *tbin = (unsigned)b; return; }
  }
  *tbin = 0u;
}

/* ---------------- compact candidates >= threshold bin ---------------- */
__global__ void k_collect(const unsigned* __restrict__ bits, const unsigned* __restrict__ tbin,
                          unsigned* __restrict__ cnt, unsigned long long* __restrict__ cand) {
  unsigned tb = *tbin;
  int stride = gridDim.x * blockDim.x;
  for (int i = blockIdx.x * blockDim.x + threadIdx.x; i < N_TOTAL; i += stride) {
    unsigned k = mono_key(bits[i]);
    if ((k >> 19) >= tb) {
      unsigned pos = atomicAdd(cnt, 1u);
      if (pos < CAP)
        cand[pos] = ((unsigned long long)k << 32) | (unsigned)(~(unsigned)i);
    }
  }
}

/* ---------------- one-WG bitonic sort, emit exact top-1000 ---------------- */
__global__ void k_sort(const unsigned long long* __restrict__ cand,
                       const unsigned* __restrict__ pcnt,
                       float* __restrict__ topscore, unsigned* __restrict__ topidx) {
  __shared__ unsigned long long sd[CAP];   /* 32 KB */
  int tid = threadIdx.x;                   /* 512 threads */
  unsigned cnt = *pcnt; if (cnt > CAP) cnt = CAP;
  for (int i = tid; i < CAP; i += 512) sd[i] = (i < (int)cnt) ? cand[i] : 0ull;
  __syncthreads();
  for (int k = 2; k <= CAP; k <<= 1) {
    for (int j = k >> 1; j > 0; j >>= 1) {
      for (int i = tid; i < CAP; i += 512) {
        int l = i ^ j;
        if (l > i) {
          unsigned long long a = sd[i], b = sd[l];
          bool dir = ((i & k) == 0);           /* descending overall */
          bool sw = dir ? (a < b) : (a > b);
          if (sw) { sd[i] = b; sd[l] = a; }
        }
      }
      __syncthreads();
    }
  }
  for (int i = tid; i < TOPK; i += 512) {
    unsigned long long e = sd[i];
    unsigned key = (unsigned)(e >> 32);
    unsigned idx = ~(unsigned)e;
    unsigned fb  = (key & 0x80000000u) ? (key ^ 0x80000000u) : ~key;
    topscore[i] = __uint_as_float(fb);       /* raw logit */
    topidx[i]   = idx;
  }
}

/* ---------------- gather boxes, offsets, areas, non-keep outputs ---------------- */
__global__ void k_prep(const float* __restrict__ box_pred,
                       const int* __restrict__ pimg_h, const int* __restrict__ pimg_w,
                       float* __restrict__ ts, const unsigned* __restrict__ ti_,
                       float* __restrict__ bx1, float* __restrict__ by1,
                       float* __restrict__ bx2, float* __restrict__ by2,
                       float* __restrict__ bar, unsigned* __restrict__ val,
                       float* __restrict__ out) {
  int tid = threadIdx.x;                     /* 1024 threads, 1 block */
  if (tid < TOPK) {
    float logit = ts[tid];
    float s = 1.0f / (1.0f + __expf(-logit));
    unsigned g = ti_[tid];
    unsigned lab = g % (unsigned)NUM_CLASSES;
    unsigned anc = g / (unsigned)NUM_CLASSES;
    float x1 = box_pred[anc * 4 + 0], y1 = box_pred[anc * 4 + 1];
    float x2 = box_pred[anc * 4 + 2], y2 = box_pred[anc * 4 + 3];
    float off = (float)lab * MAX_COORD;
    bx1[tid] = x1 + off; by1[tid] = y1 + off;
    bx2[tid] = x2 + off; by2[tid] = y2 + off;
    bar[tid] = (x2 - x1) * (y2 - y1) * AREA_SCALE;
    val[tid] = (s > CONF_THRESH) ? 1u : 0u;
    ts[tid]  = s;                            /* overwrite logit with sigmoid */
    float iw = (float)(*pimg_w), ih = (float)(*pimg_h);
    out[tid * 4 + 0] = fminf(fmaxf(x1 / iw, 0.0f), 1.0f);
    out[tid * 4 + 1] = fminf(fmaxf(y1 / ih, 0.0f), 1.0f);
    out[tid * 4 + 2] = fminf(fmaxf(x2 / iw, 0.0f), 1.0f);
    out[tid * 4 + 3] = fminf(fmaxf(y2 / ih, 0.0f), 1.0f);
    out[5000 + tid]  = (float)lab;
  } else {
    /* padding rows 1000..1023: zero-area boxes, all disjoint -> IoU == 0 */
    float base = -1.0e7f - (float)tid * 1.0e5f;
    bx1[tid] = base; by1[tid] = base; bx2[tid] = base; by2[tid] = base;
    bar[tid] = 0.0f; val[tid] = 0u;
  }
}

/* ---------------- WMMA-assisted pairwise IoU -> suppression bitmask ----------------
 * Per block: async-stage the contiguous 5x1024 f32 box/area arrays (20 KB) into LDS
 * via global_load_async_to_lds_b128 (+ s_wait_asynccnt), then one wave per 16x16
 * tile. Matrix pipe computes denom sum area[i]+area[j] (rank-2 f16 WMMA, f32
 * accum); VALU computes intersections from LDS. */
__global__ void k_iou(const float* __restrict__ gbox, unsigned* __restrict__ mask) {
  __shared__ uint4 sbuf[1280];               /* 20 KB: bx1|by1|bx2|by2|bar */
  int tid = threadIdx.x;                     /* 256 threads */
  {
    const uint4* g = (const uint4*)gbox;
#ifdef HAVE_ASYNC_LDS
    for (int k = 0; k < 5; ++k) {
      int idx = tid + k * 256;
      __builtin_amdgcn_global_load_async_to_lds_b128(
          (as1_v4i_p)(g + idx), (as3_v4i_p)(sbuf + idx), 0, 0);
    }
    __builtin_amdgcn_s_wait_asynccnt(0);
#else
    for (int k = 0; k < 5; ++k) {
      int idx = tid + k * 256;
      sbuf[idx] = g[idx];
    }
#endif
    __syncthreads();
  }
  const float* sb  = (const float*)sbuf;
  const float* bx1 = sb;
  const float* by1 = sb + 1024;
  const float* bx2 = sb + 2048;
  const float* by2 = sb + 3072;
  const float* bar = sb + 4096;

  int wid  = (blockIdx.x * blockDim.x + tid) >> 5;
  int lane = tid & 31;
  int ti = wid >> 6, tj = wid & 63;
  if (tj < ti) return;                       /* only j>i can suppress (wave-uniform) */

  int n = lane & 15;
  int j = tj * 16 + n;
  float jx1 = bx1[j], jy1 = by1[j], jx2 = bx2[j], jy2 = by2[j];
  float aI = bar[ti * 16 + n];
  float aJ = bar[j];

  bool lo = lane < 16;
  v16h A = {}; v16h B = {};
  A[0] = lo ? (_Float16)aI   : (_Float16)0.0f;   /* A[m][0] = area_i[m]   */
  A[1] = lo ? (_Float16)1.0f : (_Float16)0.0f;   /* A[m][1] = 1           */
  B[0] = lo ? (_Float16)1.0f : (_Float16)0.0f;   /* B[0][n] = 1           */
  B[1] = lo ? (_Float16)aJ   : (_Float16)0.0f;   /* B[1][n] = area_j[n]   */
  v8f C = {};
  C = __builtin_amdgcn_wmma_f32_16x16x32_f16(false, A, false, B,
                                             (short)0, C, false, false);
  /* D[m][n] = area_i[m] + area_j[n] (scaled); lanes 0-15: M=r, lanes 16-31: M=r+8 */
  int mbase = ti * 16 + (lo ? 0 : 8);
  for (int r = 0; r < 8; ++r) {
    int i = mbase + r;
    float ix1 = bx1[i], iy1 = by1[i], ix2 = bx2[i], iy2 = by2[i];
    float xx1 = fmaxf(ix1, jx1), yy1 = fmaxf(iy1, jy1);
    float xx2 = fminf(ix2, jx2), yy2 = fminf(iy2, jy2);
    float inter = fmaxf(xx2 - xx1, 0.0f) * fmaxf(yy2 - yy1, 0.0f);
    float den = C[r] * AREA_UNSCALE - inter + 1e-9f;
    bool pred = (inter > NMS_THRESH * den) && (j > i);
    unsigned bal = (unsigned)__ballot(pred);
    if (lane < 2) {
      unsigned bits = (lane == 0) ? (bal & 0xFFFFu) : (bal >> 16);
      int row  = ti * 16 + r + ((lane == 0) ? 0 : 8);
      int word = tj >> 1;
      int sh   = (tj & 1) * 16;
      if (bits) atomicOr(&mask[row * 32 + word], bits << sh);
    }
  }
}

/* ---------------- serial NMS scan: one wave, remv word per lane ---------------- */
__global__ void k_nms(const unsigned* __restrict__ mask, const unsigned* __restrict__ val,
                      const float* __restrict__ s, float* __restrict__ out) {
  int lane = threadIdx.x;                    /* 32 threads */
  unsigned remv = 0u;
  for (int i = 0; i < TOPK; ++i) {
    unsigned w = __shfl(remv, i >> 5);
    bool sup = (w >> (i & 31)) & 1u;
    bool k = (val[i] != 0u) && !sup;
    if (k) remv |= mask[i * 32 + lane];
    if (lane == 0) {
      out[6000 + i] = k ? 1.0f : 0.0f;
      out[4000 + i] = k ? s[i] : 0.0f;
    }
  }
}

extern "C" void kernel_launch(void* const* d_in, const int* in_sizes, int n_in,
                              void* d_out, int out_size, void* d_ws, size_t ws_size,
                              hipStream_t stream) {
  const float* cls = (const float*)d_in[0];
  const float* box = (const float*)d_in[1];
  const int* ph = (const int*)d_in[2];
  const int* pw = (const int*)d_in[3];
  float* out = (float*)d_out;
  unsigned* ws = (unsigned*)d_ws;
  unsigned long long* cand = (unsigned long long*)(ws + OFF_CAND);

  k_zero<<<256, 256, 0, stream>>>(ws, WS_WORDS);
  k_hist<<<2048, 256, 0, stream>>>((const uint4*)cls, ws + OFF_HIST, N4_TOTAL);
  k_thresh<<<1, 1, 0, stream>>>(ws + OFF_HIST, ws + OFF_TBIN);
  k_collect<<<2048, 256, 0, stream>>>((const unsigned*)cls, ws + OFF_TBIN,
                                      ws + OFF_CNT, cand);
  k_sort<<<1, 512, 0, stream>>>(cand, ws + OFF_CNT,
                                (float*)(ws + OFF_TS), ws + OFF_TI);
  k_prep<<<1, 1024, 0, stream>>>(box, ph, pw,
                                 (float*)(ws + OFF_TS), ws + OFF_TI,
                                 (float*)(ws + OFF_BX1), (float*)(ws + OFF_BY1),
                                 (float*)(ws + OFF_BX2), (float*)(ws + OFF_BY2),
                                 (float*)(ws + OFF_BAR), ws + OFF_VAL, out);
  k_iou<<<512, 256, 0, stream>>>((const float*)(ws + OFF_BX1), ws + OFF_MASK);
  k_nms<<<1, 32, 0, stream>>>(ws + OFF_MASK, ws + OFF_VAL,
                              (const float*)(ws + OFF_TS), out);
}
